// LATTE_75222057222535
// MI455X (gfx1250) — matrix operand order repaired and checked by reading, not compiled, not run on hardware
//
#include <hip/hip_runtime.h>
#include <math.h>

#define NN   100000   // nodes
#define DIN  256      // input dim
#define DOUT 128      // output dim
#define NHD  4        // heads
#define NC   32       // channels per head
#define NR   3        // relations (metapaths)
#define NE   600000   // edges per relation

typedef __attribute__((ext_vector_type(16))) _Float16 v16h;
typedef __attribute__((ext_vector_type(8)))  float    v8f;

// ---------- order-preserving float<->uint key for atomicMax on floats ----------
__device__ __forceinline__ unsigned f2key(float f) {
    int i = __float_as_int(f);
    return (i >= 0) ? (unsigned(i) | 0x80000000u) : ~unsigned(i);
}
__device__ __forceinline__ float key2f(unsigned k) {
    int i = (k & 0x80000000u) ? int(k & 0x7fffffffu) : int(~k);
    return __int_as_float(i);
}

// ---------- zero fill (emb + max-keys + sums), float4 per thread ----------
__global__ void fill_zero4(float4* __restrict__ p, long n4) {
    long i = (long)blockIdx.x * blockDim.x + threadIdx.x;
    if (i < n4) p[i] = make_float4(0.f, 0.f, 0.f, 0.f);
}

// ---------- pre-pack Wl/Wr into per-lane WMMA B-operand layout, f16 ----------
// wpk index = ((Wsel*8 + ntile)*8 + kchunk)*32 + lane, 16 halves each (32 B/lane).
// Element e of a lane's v16h = W[kchunk*32 + (lane>>4)*16 + e][ntile*16 + (lane&15)].
__global__ void pack_w(const float* __restrict__ Wl, const float* __restrict__ Wr,
                       _Float16* __restrict__ wpk)
{
    int t = blockIdx.x * blockDim.x + threadIdx.x;   // 0 .. 4095
    if (t >= 2 * 2048) return;
    const float* W = (t < 2048) ? Wl : Wr;
    const int tt   = t & 2047;
    const int nt   = tt >> 8;          // 0..7  n-tile
    const int kc   = (tt >> 5) & 7;    // 0..7  k-chunk
    const int lane = tt & 31;
    const int n    = nt * 16 + (lane & 15);
    const int kb   = kc * 32 + (lane >> 4) * 16;
    _Float16* dst = wpk + (size_t)t * 16;
    #pragma unroll
    for (int e = 0; e < 16; ++e)
        dst[e] = (_Float16)W[(size_t)(kb + e) * DOUT + n];
}

// ---------- projection GEMM: out = relu(x @ W + b), via v_wmma_f32_16x16x32_f16 ----------
// grid.x = NN/16 row tiles, grid.y = {0:l, 1:r}; block = 256 (8 waves, wave w -> n-tile w)
__global__ __launch_bounds__(256) void proj_wmma(
    const float* __restrict__ x,
    const _Float16* __restrict__ wpk,
    const float* __restrict__ bl, const float* __restrict__ br,
    float* __restrict__ lbuf, float* __restrict__ rbuf)
{
    const int wave  = threadIdx.x >> 5;
    const int lane  = threadIdx.x & 31;
    const int hf    = lane >> 4;      // lane half (0/1)
    const int l16   = lane & 15;
    const int row0  = blockIdx.x * 16;
    const int n0    = wave * 16;

    const float* bias = blockIdx.y ? br : bl;
    float*       out  = blockIdx.y ? rbuf : lbuf;
    // pre-packed B tiles for this (matrix, ntile): 8 k-chunks x 32 lanes x 16 halves
    const _Float16* wbase = wpk
        + (((size_t)blockIdx.y * 8 + wave) * 8) * (32 * 16)
        + (size_t)lane * 16;

    v8f acc = {};
    const float* xrow = x + (size_t)(row0 + l16) * DIN;

    #pragma unroll
    for (int kc = 0; kc < 8; ++kc) {
        // A tile 16x32 f16: lane half selects K offset; elems 0..15 -> K+16*hf+0..15
        const float* xp = xrow + kc * 32 + hf * 16;
        v16h a;
        #pragma unroll
        for (int j = 0; j < 8; ++j) a[j]     = (_Float16)xp[j];
        #pragma unroll
        for (int j = 0; j < 8; ++j) a[8 + j] = (_Float16)xp[8 + j];
        // B tile: one 32-byte contiguous per-lane load from the packed buffer
        v16h b = *(const v16h*)(wbase + (size_t)kc * (32 * 16));

        acc = __builtin_amdgcn_wmma_f32_16x16x32_f16(
            false, a, false, b, (short)0, acc, false, false);
    }

    const float bb = bias[n0 + l16];
    #pragma unroll
    for (int v = 0; v < 8; ++v) {
        const int row = row0 + v + hf * 8;       // C/D layout: lanes16-31 -> M = v+8
        float val = acc[v] + bb;
        val = val > 0.f ? val : 0.f;
        out[(size_t)row * DOUT + n0 + l16] = val;
    }
}

// ---------- pass A: per-(edge,head) attention logits + segment max ----------
__global__ void edge_logits(const int* __restrict__ ei,
                            const float* __restrict__ lbuf,
                            const float* __restrict__ rbuf,
                            const float* __restrict__ attn,
                            float* __restrict__ logits,
                            unsigned* __restrict__ mbuf)
{
    const int rel = blockIdx.y;
    long idx = (long)blockIdx.x * blockDim.x + threadIdx.x;
    if (idx >= (long)NE * NHD) return;
    const int e = (int)(idx >> 2);
    const int h = (int)(idx & 3);
    const int src = ei[(size_t)rel * 2 * NE + e];
    const int dst = ei[(size_t)rel * 2 * NE + NE + e];

    const float4* lp = (const float4*)(lbuf + (size_t)src * DOUT + h * NC);
    const float4* rp = (const float4*)(rbuf + (size_t)dst * DOUT + h * NC);
    const float4* al = (const float4*)(attn + ((size_t)rel * NHD + h) * (2 * NC));
    const float4* ar = al + NC / 4;

    float a = 0.f;
    #pragma unroll
    for (int c = 0; c < NC / 4; ++c) {
        float4 lv = lp[c], av = al[c];
        a += lv.x * av.x + lv.y * av.y + lv.z * av.z + lv.w * av.w;
        float4 rv = rp[c], bv = ar[c];
        a += rv.x * bv.x + rv.y * bv.y + rv.z * bv.z + rv.w * bv.w;
    }
    a = a > 0.f ? a : 0.2f * a;   // leaky relu
    logits[(size_t)rel * NE * NHD + idx] = a;
    atomicMax(&mbuf[((size_t)rel * NN + dst) * NHD + h], f2key(a));
}

// ---------- pass B: e = exp(a - max), segment sum ----------
__global__ void edge_expsum(const int* __restrict__ ei,
                            float* __restrict__ logits,
                            const unsigned* __restrict__ mbuf,
                            float* __restrict__ sbuf)
{
    const int rel = blockIdx.y;
    long idx = (long)blockIdx.x * blockDim.x + threadIdx.x;
    if (idx >= (long)NE * NHD) return;
    const int e = (int)(idx >> 2);
    const int h = (int)(idx & 3);
    const int dst = ei[(size_t)rel * 2 * NE + NE + e];
    const float m = key2f(mbuf[((size_t)rel * NN + dst) * NHD + h]);
    const size_t li = (size_t)rel * NE * NHD + idx;
    const float ev = __expf(logits[li] - m);
    logits[li] = ev;
    atomicAdd(&sbuf[((size_t)rel * NN + dst) * NHD + h], ev);
}

// ---------- pass C: alpha-weighted scatter of l[src] into emb[rel][dst] ----------
__global__ void edge_scatter(const int* __restrict__ ei,
                             const float* __restrict__ logits,
                             const float* __restrict__ sbuf,
                             const float* __restrict__ lbuf,
                             float* __restrict__ emb)
{
    const int rel = blockIdx.y;
    long idx = (long)blockIdx.x * blockDim.x + threadIdx.x;
    if (idx >= (long)NE * NHD) return;
    const int e = (int)(idx >> 2);
    const int h = (int)(idx & 3);
    const int src = ei[(size_t)rel * 2 * NE + e];
    const int dst = ei[(size_t)rel * 2 * NE + NE + e];
    const float ev = logits[(size_t)rel * NE * NHD + idx];
    const float sv = sbuf[((size_t)rel * NN + dst) * NHD + h];
    const float alpha = ev / (sv + 1e-16f);

    const float4* lp = (const float4*)(lbuf + (size_t)src * DOUT + h * NC);
    float* ep = emb + ((size_t)rel * NN + dst) * DOUT + h * NC;
    #pragma unroll
    for (int c = 0; c < NC / 4; ++c) {
        float4 lv = lp[c];
        atomicAdd(&ep[4 * c + 0], lv.x * alpha);
        atomicAdd(&ep[4 * c + 1], lv.y * alpha);
        atomicAdd(&ep[4 * c + 2], lv.z * alpha);
        atomicAdd(&ep[4 * c + 3], lv.w * alpha);
    }
}

// ---------- final: relation-level beta softmax + combine, one thread per (node, head) ----------
__global__ void combine(const float* __restrict__ lbuf,
                        const float* __restrict__ rbuf,
                        const float* __restrict__ emb,
                        const float* __restrict__ ral,
                        const float* __restrict__ rar,
                        const float* __restrict__ rbias,
                        float* __restrict__ out)
{
    long idx = (long)blockIdx.x * blockDim.x + threadIdx.x;
    if (idx >= (long)NN * NHD) return;
    const int n = (int)(idx >> 2);
    const int h = (int)(idx & 3);

    const float* rp  = rbuf + (size_t)n * DOUT + h * NC;
    const float* alp = ral + h * NC;
    float blv[NC];
    #pragma unroll
    for (int c = 0; c < NC; ++c) {
        float v = rp[c] * alp[c];
        blv[c] = v > 0.f ? v : 0.f;
    }

    const float* eps[NR + 1];
    eps[0] = emb + ((size_t)0 * NN + n) * DOUT + h * NC;
    eps[1] = emb + ((size_t)1 * NN + n) * DOUT + h * NC;
    eps[2] = emb + ((size_t)2 * NN + n) * DOUT + h * NC;
    eps[3] = lbuf + (size_t)n * DOUT + h * NC;   // self slot

    float beta[NR + 1];
    #pragma unroll
    for (int rr = 0; rr < NR + 1; ++rr) {
        const float* ep = eps[rr];
        const float* ap = rar + ((size_t)rr * NHD + h) * NC;
        float d = 0.f;
        #pragma unroll
        for (int c = 0; c < NC; ++c) {
            float v = ep[c] * ap[c];
            v = v > 0.f ? v : 0.f;
            d += blv[c] * v;
        }
        beta[rr] = d + rbias[rr];
    }

    float mx = beta[0];
    #pragma unroll
    for (int rr = 1; rr < NR + 1; ++rr) mx = fmaxf(mx, beta[rr]);
    float w[NR + 1], s = 0.f;
    #pragma unroll
    for (int rr = 0; rr < NR + 1; ++rr) { w[rr] = __expf(beta[rr] - mx); s += w[rr]; }
    const float inv = 1.f / s;
    #pragma unroll
    for (int rr = 0; rr < NR + 1; ++rr) w[rr] *= inv;

    float* op = out + (size_t)n * DOUT + h * NC;
    #pragma unroll
    for (int c = 0; c < NC; ++c) {
        float o = w[0] * eps[0][c] + w[1] * eps[1][c] + w[2] * eps[2][c] + w[3] * eps[3][c];
        op[c] = o > 0.f ? o : 0.f;
    }
}

extern "C" void kernel_launch(void* const* d_in, const int* in_sizes, int n_in,
                              void* d_out, int out_size, void* d_ws, size_t ws_size,
                              hipStream_t stream)
{
    (void)in_sizes; (void)n_in; (void)out_size; (void)ws_size;
    const float* x     = (const float*)d_in[0];
    const int*   ei    = (const int*)  d_in[1];
    const float* Wl    = (const float*)d_in[2];
    const float* bl    = (const float*)d_in[3];
    const float* Wr    = (const float*)d_in[4];
    const float* br    = (const float*)d_in[5];
    const float* attn  = (const float*)d_in[6];
    const float* ral   = (const float*)d_in[7];
    const float* rar   = (const float*)d_in[8];
    const float* rbias = (const float*)d_in[9];
    float* out = (float*)d_out;
    float* ws  = (float*)d_ws;

    // workspace layout (floats); emb/m/s contiguous so one fill zeroes all accumulators
    size_t off = 0;
    float*    lbuf   = ws + off; off += (size_t)NN * DOUT;        // 12.8M
    float*    rbuf   = ws + off; off += (size_t)NN * DOUT;        // 12.8M
    float*    emb    = ws + off; off += (size_t)NR * NN * DOUT;   // 38.4M
    unsigned* mbuf   = (unsigned*)(ws + off); off += (size_t)NR * NN * NHD; // 1.2M
    float*    sbuf   = ws + off; off += (size_t)NR * NN * NHD;    // 1.2M
    float*    logits = ws + off; off += (size_t)NR * NE * NHD;    // 7.2M
    _Float16* wpk    = (_Float16*)(ws + off); off += 32768;       // 2*2048*16 halves = 128KB

    const size_t zfloats = (size_t)NR * NN * DOUT + 2 * (size_t)NR * NN * NHD;
    const size_t n4 = zfloats / 4;
    fill_zero4<<<(unsigned)((n4 + 255) / 256), 256, 0, stream>>>((float4*)emb, (long)n4);

    pack_w<<<(2 * 2048 + 255) / 256, 256, 0, stream>>>(Wl, Wr, wpk);

    dim3 gproj(NN / 16, 2);
    proj_wmma<<<gproj, 256, 0, stream>>>(x, wpk, bl, br, lbuf, rbuf);

    const long eh = (long)NE * NHD;
    dim3 gedge((unsigned)((eh + 255) / 256), NR);
    edge_logits <<<gedge, 256, 0, stream>>>(ei, lbuf, rbuf, attn, logits, mbuf);
    edge_expsum <<<gedge, 256, 0, stream>>>(ei, logits, mbuf, sbuf);
    edge_scatter<<<gedge, 256, 0, stream>>>(ei, logits, sbuf, lbuf, emb);

    const long nh = (long)NN * NHD;
    combine<<<(unsigned)((nh + 255) / 256), 256, 0, stream>>>(lbuf, rbuf, emb, ral, rar, rbias, out);
}
